// CCALoss_89111981457697
// MI455X (gfx1250) — 8-bit matrix operand order repaired and checked by reading, compiled, not run on hardware
//
#include <hip/hip_runtime.h>
#include <hip/hip_bf16.h>
#include <math.h>

typedef __attribute__((ext_vector_type(8))) int v8i;

#define B_DIM 512
#define C_DIM 256

__device__ __forceinline__ float waveReduceSum(float v) {
    for (int o = 16; o > 0; o >>= 1) v += __shfl_xor(v, o, 32);
    return v;
}
__device__ __forceinline__ float waveReduceMax(float v) {
    for (int o = 16; o > 0; o >>= 1) v = fmaxf(v, __shfl_xor(v, o, 32));
    return v;
}
__device__ __forceinline__ float blockReduceSum(float v, float* sh) {
    int wave = threadIdx.x >> 5, lane = threadIdx.x & 31;
    v = waveReduceSum(v);
    if (lane == 0) sh[wave] = v;
    __syncthreads();
    if (wave == 0) {
        float t = (lane < 8) ? sh[lane] : 0.f;
        t = waveReduceSum(t);
        if (lane == 0) sh[0] = t;
    }
    __syncthreads();
    v = sh[0];
    __syncthreads();
    return v;
}
__device__ __forceinline__ float blockReduceMax(float v, float* sh) {
    int wave = threadIdx.x >> 5, lane = threadIdx.x & 31;
    v = waveReduceMax(v);
    if (lane == 0) sh[wave] = v;
    __syncthreads();
    if (wave == 0) {
        float t = (lane < 8) ? sh[lane] : -INFINITY;
        t = waveReduceMax(t);
        if (lane == 0) sh[0] = t;
    }
    __syncthreads();
    v = sh[0];
    __syncthreads();
    return v;
}
__device__ __forceinline__ float rowLse512(const float* row, float* sh) {
    float a = row[threadIdx.x];
    float b = row[threadIdx.x + 256];
    float m = blockReduceMax(fmaxf(a, b), sh);
    float s = blockReduceSum(__expf(a - m) + __expf(b - m), sh);
    return m + __logf(s);
}

__global__ void cca_zero_kernel(float* gAcc) {
    if (threadIdx.x < 8) gAcc[threadIdx.x] = 0.f;
}

__global__ void cca_prep_kernel(const float* __restrict__ imgL,
                                const float* __restrict__ txtL,
                                const float* __restrict__ cLogits,
                                const int*   __restrict__ med,
                                unsigned char* __restrict__ wsW,
                                int* __restrict__ wsRS,
                                float* __restrict__ gAcc) {
    __shared__ float sh[8];
    const int i = blockIdx.x;
    const int c = threadIdx.x;

    int   mc    = med[i * C_DIM + c];
    float x     = cLogits[i * C_DIM + c];
    float maskf = (mc != -1) ? 1.f : 0.f;
    float tgt   = (mc == 1) ? 1.f : 0.f;
    float sp  = fmaxf(x, 0.f) + log1pf(__expf(-fabsf(x)));
    float bce = (sp - x * tgt) * maskf;

    unsigned char w = (mc == 1) ? (unsigned char)1 : (unsigned char)0;
    wsW[i * C_DIM + c] = w;

    float bceSum  = blockReduceSum(bce, sh);
    float maskSum = blockReduceSum(maskf, sh);
    float rsum    = blockReduceSum((float)w, sh);

    float lseImg = rowLse512(imgL + (size_t)i * B_DIM, sh);
    float lseTxt = rowLse512(txtL + (size_t)i * B_DIM, sh);

    if (c == 0) {
        wsRS[i] = (int)(rsum + 0.5f);
        atomicAdd(&gAcc[0], lseImg - imgL[(size_t)i * B_DIM + i]);
        atomicAdd(&gAcc[1], lseTxt - txtL[(size_t)i * B_DIM + i]);
        atomicAdd(&gAcc[2], bceSum);
        atomicAdd(&gAcc[3], maskSum);
    }
}

__global__ void cca_gram_kl_kernel(const float* __restrict__ csim,
                                   const unsigned char* __restrict__ wsW,
                                   const int* __restrict__ wsRS,
                                   float* __restrict__ gAcc) {
    __shared__ float simS[16 * B_DIM];
    const uint32_t* W32 = (const uint32_t*)wsW;

    const int tid  = threadIdx.x;
    const int wave = tid >> 5;
    const int lane = tid & 31;
    const int half = lane >> 4;
    const int l16  = lane & 15;
    const int m0   = blockIdx.x * 16;

    v8i A[4];
    {
        const uint32_t* row = W32 + (size_t)(m0 + l16) * 64;
        #pragma unroll
        for (int kb = 0; kb < 4; ++kb) {
            const int wb = kb * 16;
            #pragma unroll
            for (int g = 0; g < 4; ++g) {
                uint2 p = *(const uint2*)(row + wb + (g >> 1) * 8 + (g & 1) * 2 + half * 4);
                A[kb][2 * g]     = (int)p.x;
                A[kb][2 * g + 1] = (int)p.y;
            }
        }
    }
    int rsM[8];
    #pragma unroll
    for (int r = 0; r < 8; ++r) rsM[r] = wsRS[m0 + r + 8 * half];

    for (int q = 0; q < 4; ++q) {
        const int n0 = (wave + q * 8) * 16;
        const uint32_t* brow = W32 + (size_t)(n0 + l16) * 64;
        const int rsN = wsRS[n0 + l16];

        v8i accv = {0, 0, 0, 0, 0, 0, 0, 0};
        #pragma unroll
        for (int kb = 0; kb < 4; ++kb) {
            uint4 p0 = *(const uint4*)(brow + kb * 16 + half * 4);
            uint4 p1 = *(const uint4*)(brow + kb * 16 + 8 + half * 4);
            v8i Bv;
            Bv[0] = (int)p0.x; Bv[1] = (int)p0.y; Bv[2] = (int)p0.z; Bv[3] = (int)p0.w;
            Bv[4] = (int)p1.x; Bv[5] = (int)p1.y; Bv[6] = (int)p1.z; Bv[7] = (int)p1.w;
            accv = __builtin_amdgcn_wmma_i32_16x16x64_iu8(
                 false, A[kb],  false, Bv, accv,
                 false,  false);
        }
        #pragma unroll
        for (int r = 0; r < 8; ++r) {
            int inter = accv[r];
            int uni   = rsM[r] + rsN - inter;
            float s   = (float)inter * __builtin_amdgcn_rcpf((float)uni);
            s         = (uni > 0) ? s : 0.f;
            simS[(r + 8 * half) * B_DIM + n0 + l16] = s;
        }
    }
    __syncthreads();

    const float invT = 1.0f / 0.07f;
    float klPart = 0.f;
    #pragma unroll
    for (int rr = 0; rr < 2; ++rr) {
        const int lr = wave * 2 + rr;
        const int gi = m0 + lr;
        float xs[16], cs[16];
        float mx = -INFINITY, mcv = -INFINITY;
        #pragma unroll
        for (int t = 0; t < 16; ++t) {
            const int j = lane + 32 * t;
            xs[t] = simS[lr * B_DIM + j] * invT;
            cs[t] = csim[(size_t)gi * B_DIM + j];
            mx  = fmaxf(mx, xs[t]);
            mcv = fmaxf(mcv, cs[t]);
        }
        mx  = waveReduceMax(mx);
        mcv = waveReduceMax(mcv);
        float sx = 0.f, sc = 0.f;
        #pragma unroll
        for (int t = 0; t < 16; ++t) { sx += __expf(xs[t] - mx); sc += __expf(cs[t] - mcv); }
        sx = waveReduceSum(sx);
        sc = waveReduceSum(sc);
        const float lseX = mx + __logf(sx);
        const float lseC = mcv + __logf(sc);
        float kl = 0.f;
        #pragma unroll
        for (int t = 0; t < 16; ++t) {
            float lt = xs[t] - lseX;
            float lp = cs[t] - lseC;
            kl += __expf(lt) * (lt - lp);
        }
        kl = waveReduceSum(kl);
        if (lane == 0) klPart += kl;
    }
    if (lane == 0) atomicAdd(&gAcc[4], klPart);
}

__global__ void cca_finalize_kernel(const float* __restrict__ gAcc,
                                    float* __restrict__ out) {
    const float img     = gAcc[0] / (float)B_DIM;
    const float txt     = gAcc[1] / (float)B_DIM;
    const float clip    = 0.5f * (img + txt);
    const float concept = gAcc[2] / (gAcc[3] + 1e-8f);
    const float kl      = gAcc[4] / (float)B_DIM;
    out[0] = clip + 0.5f * concept + 0.3f * kl;
}

extern "C" void kernel_launch(void* const* d_in, const int* in_sizes, int n_in,
                              void* d_out, int out_size, void* d_ws, size_t ws_size,
                              hipStream_t stream) {
    const float* imgL  = (const float*)d_in[0];
    const float* txtL  = (const float*)d_in[1];
    const float* cLog  = (const float*)d_in[2];
    const float* csim  = (const float*)d_in[3];
    const int*   med   = (const int*)  d_in[4];

    unsigned char* wsW  = (unsigned char*)d_ws;
    int*           wsRS = (int*)  ((char*)d_ws + B_DIM * C_DIM);
    float*         gAcc = (float*)((char*)d_ws + B_DIM * C_DIM + B_DIM * 4);

    cca_zero_kernel<<<1, 32, 0, stream>>>(gAcc);
    cca_prep_kernel<<<B_DIM, 256, 0, stream>>>(imgL, txtL, cLog, med, wsW, wsRS, gAcc);
    cca_gram_kl_kernel<<<B_DIM / 16, 256, 0, stream>>>(csim, wsW, wsRS, gAcc);
    cca_finalize_kernel<<<1, 1, 0, stream>>>(gAcc, (float*)d_out);
}
